// GCN_Layer_80264348828246
// MI455X (gfx1250) — matrix operand order repaired and checked by reading, not compiled, not run on hardware
//
#include <hip/hip_runtime.h>
#include <hip/hip_bf16.h>

constexpr int N = 100000;
constexpr int E = 1600000;
constexpr int C = 128;
constexpr int L = 4;
constexpr int LC = L * C;
constexpr float EPSV = 1e-5f;
constexpr int RBLK = 512;

typedef __attribute__((ext_vector_type(16))) _Float16 v16h;
typedef __attribute__((ext_vector_type(8)))  float    v8f;

// ---------------- degree / normalization ----------------
__global__ void k_zero_f(float* p, int n) {
  int i = blockIdx.x * 256 + threadIdx.x;
  if (i < n) p[i] = 0.f;
}

__global__ void k_deg(const int* __restrict__ dst, float* __restrict__ deg) {
  int e = blockIdx.x * 256 + threadIdx.x;
  if (e < E) atomicAdd(&deg[dst[e]], 1.0f);
}

__global__ void k_dinv(float* d) {
  int i = blockIdx.x * 256 + threadIdx.x;
  if (i < N) d[i] = rsqrtf(d[i] + 1.0f);   // deg includes self-loop, always > 0
}

// ---------------- weight fp32 -> f16 pre-swizzled B fragments ----------------
// B layout used here (contiguous k, same as A; not AMD's table order): lane l holds col n=l%16; halves j=0..15
// map to K = 16*(l/16) + j within each 32-K block. Stored so each lane reads
// 16 contiguous halves: index = ((kb*8 + ntile)*32 + lane)*16 + j.
__global__ void k_convw(const float* __restrict__ W, _Float16* __restrict__ dstW) {
  int idx = blockIdx.x * 256 + threadIdx.x;       // 16384 = 128*128
  int k = idx >> 7, n = idx & 127;
  int kb = k >> 5, kr = k & 31;
  int lane = ((kr >> 4) << 4) | (n & 15);
  int j = kr & 15;
  int ntile = n >> 4;
  dstW[((((kb * 8 + ntile) * 32 + lane)) << 4) + j] = (_Float16)W[idx];
}

// ---------------- WMMA GEMM: Y = X[ N x 128 ] * W[128 x 128] (+ epilogues) ---
// MODE 0: Y = X@W                  (gcn, bias handled later with self-loop)
// MODE 1: Y = relu(LN(X@W + b))    (mlp first linear, fused LayerNorm)
// MODE 2: Y = X@W + b  -> strided output (row stride ldY, col offset colOff)
template <int MODE>
__global__ __launch_bounds__(256) void k_gemm(const float* __restrict__ X,
                                              const _Float16* __restrict__ Bw,
                                              const float* __restrict__ bias,
                                              const float* __restrict__ lng,
                                              const float* __restrict__ lnb,
                                              float* __restrict__ Y,
                                              int ldY, int colOff) {
  __shared__ __align__(16) _Float16 As[16 * 128];
  __shared__ __align__(16) float Ot[MODE == 1 ? 16 * 128 : 1];
  __shared__ float rmean[16], ristd[16];

  const int t = threadIdx.x;
  const int wv = t >> 5, lane = t & 31;
  const int lo = lane & 15, hi = lane >> 4;
  const int m0 = blockIdx.x * 16;

  { // stage 16x128 fp32 tile -> f16 LDS (coalesced, 8 elems/thread)
    int row = t >> 4;
    int col = (t & 15) * 8;
    const float4* xp = (const float4*)(X + (size_t)(m0 + row) * C + col);
    float4 f0 = xp[0], f1 = xp[1];
    union { _Float16 h[8]; uint4 q; } tq;
    tq.h[0] = (_Float16)f0.x; tq.h[1] = (_Float16)f0.y;
    tq.h[2] = (_Float16)f0.z; tq.h[3] = (_Float16)f0.w;
    tq.h[4] = (_Float16)f1.x; tq.h[5] = (_Float16)f1.y;
    tq.h[6] = (_Float16)f1.z; tq.h[7] = (_Float16)f1.w;
    *(uint4*)&As[row * 128 + col] = tq.q;
  }
  __syncthreads();

  v8f acc = {};
  const int n0 = wv * 16;
#pragma unroll
  for (int kb = 0; kb < 4; ++kb) {
    union { v16h v; uint4 q[2]; } Af, Bf;
    // A 16x32 f16 fragment, contiguous k order used here: lane l holds row l%16, j0..15 -> K=base+16*(l/16)+j
    int abase = lo * 128 + kb * 32 + hi * 16;
    Af.q[0] = *(const uint4*)&As[abase];
    Af.q[1] = *(const uint4*)&As[abase + 8];
    const uint4* bp = (const uint4*)(Bw + (size_t)(((kb * 8 + wv) * 32 + lane) << 4));
    Bf.q[0] = bp[0];
    Bf.q[1] = bp[1];
    acc = __builtin_amdgcn_wmma_f32_16x16x32_f16(false, Af.v, false, Bf.v,
                                                 (short)0, acc, false, false);
  }

  const int col = n0 + lo;          // 0..127 within the strip
  const int row0 = hi * 8;          // C/D layout: VGPR r -> row r + 8*(lane/16)

  if constexpr (MODE == 0) {
#pragma unroll
    for (int r = 0; r < 8; ++r)
      Y[(size_t)(m0 + row0 + r) * C + col] = acc[r];
  } else if constexpr (MODE == 2) {
    float bv = bias[col];
#pragma unroll
    for (int r = 0; r < 8; ++r)
      Y[(size_t)(m0 + row0 + r) * ldY + colOff + col] = acc[r] + bv;
  } else { // MODE 1: bias -> LayerNorm(row) -> ReLU
    float bv = bias[col];
#pragma unroll
    for (int r = 0; r < 8; ++r)
      Ot[(row0 + r) * 128 + col] = acc[r] + bv;
    __syncthreads();
    if (t < 16) {
      float s = 0.f, s2 = 0.f;
#pragma unroll 4
      for (int c = 0; c < 128; ++c) {
        float v = Ot[t * 128 + c];
        s += v; s2 += v * v;
      }
      float m = s * (1.0f / 128.0f);
      float var = s2 * (1.0f / 128.0f) - m * m;
      rmean[t] = m;
      ristd[t] = rsqrtf(var + EPSV);
    }
    __syncthreads();
    int rr = t >> 4;
    int c0 = (t & 15) * 8;
    float rm = rmean[rr], ri = ristd[rr];
#pragma unroll
    for (int i = 0; i < 8; ++i) {
      int c = c0 + i;
      float v = (Ot[rr * 128 + c] - rm) * ri * lng[c] + lnb[c];
      Y[(size_t)(m0 + rr) * C + c] = fmaxf(v, 0.f);
    }
  }
}

// ---------------- aggregation ----------------
__global__ void k_agg_init(const float* __restrict__ hw, const float* __restrict__ dinv,
                           const float* __restrict__ bias, float* __restrict__ agg) {
  int i4 = blockIdx.x * 256 + threadIdx.x;   // over N*C/4 float4
  if (i4 >= N * (C / 4)) return;
  int n = i4 >> 5;
  int c4 = (i4 & 31) * 4;
  float dv = dinv[n];
  float sn = dv * dv;                         // self-loop weight
  float4 v = ((const float4*)hw)[i4];
  float4 b = *(const float4*)(bias + c4);
  float4 o;
  o.x = fmaf(v.x, sn, b.x); o.y = fmaf(v.y, sn, b.y);
  o.z = fmaf(v.z, sn, b.z); o.w = fmaf(v.w, sn, b.w);
  ((float4*)agg)[i4] = o;
}

__global__ void k_scatter(const float* __restrict__ hw, const float* __restrict__ dinv,
                          const int* __restrict__ src, const int* __restrict__ dst,
                          float* __restrict__ agg) {
  int gid = blockIdx.x * 256 + threadIdx.x;   // 32 lanes per edge, 4 ch each
  int e = gid >> 5;
  if (e >= E) return;
  int ln = gid & 31;
  int s = src[e], d = dst[e];
  float nrm = dinv[s] * dinv[d];
  float4 v = *(const float4*)(hw + (size_t)s * C + ln * 4);
  float* ap = agg + (size_t)d * C + ln * 4;
  atomicAdd(ap + 0, v.x * nrm);
  atomicAdd(ap + 1, v.y * nrm);
  atomicAdd(ap + 2, v.z * nrm);
  atomicAdd(ap + 3, v.w * nrm);
}

// ---------------- batch norm (two-stage deterministic reduction) -------------
__global__ void k_bn_partial(const float* __restrict__ X, float* __restrict__ part) {
  int t = threadIdx.x;
  int c = t & 127;
  int h = t >> 7;                             // 2 rows per block per step
  float s = 0.f, s2 = 0.f;
  for (int r = blockIdx.x * 2 + h; r < N; r += gridDim.x * 2) {
    float v = X[(size_t)r * C + c];
    s += v; s2 += v * v;
  }
  __shared__ float sh[512];
  sh[t] = s; sh[256 + t] = s2;
  __syncthreads();
  if (t < 128) {
    part[blockIdx.x * 256 + c]       = sh[c] + sh[128 + c];
    part[blockIdx.x * 256 + 128 + c] = sh[256 + c] + sh[384 + c];
  }
}

__global__ void k_bn_finalize(const float* __restrict__ part, float* __restrict__ mu,
                              float* __restrict__ istd, int nblk) {
  int c = threadIdx.x;                        // 128 threads
  float s = 0.f, s2 = 0.f;
  for (int b = 0; b < nblk; ++b) { s += part[b * 256 + c]; s2 += part[b * 256 + 128 + c]; }
  float m = s / (float)N;
  float var = s2 / (float)N - m * m;
  mu[c] = m;
  istd[c] = rsqrtf(var + EPSV);
}

__global__ void k_bn_apply(const float* __restrict__ X, const float* __restrict__ mu,
                           const float* __restrict__ istd, const float* __restrict__ g,
                           const float* __restrict__ b, float* __restrict__ Y) {
  int i4 = blockIdx.x * 256 + threadIdx.x;
  if (i4 >= N * (C / 4)) return;
  int c4 = (i4 & 31) * 4;
  float4 v = ((const float4*)X)[i4];
  float4 o;
  o.x = fmaxf((v.x - mu[c4 + 0]) * istd[c4 + 0] * g[c4 + 0] + b[c4 + 0], 0.f);
  o.y = fmaxf((v.y - mu[c4 + 1]) * istd[c4 + 1] * g[c4 + 1] + b[c4 + 1], 0.f);
  o.z = fmaxf((v.z - mu[c4 + 2]) * istd[c4 + 2] * g[c4 + 2] + b[c4 + 2], 0.f);
  o.w = fmaxf((v.w - mu[c4 + 3]) * istd[c4 + 3] * g[c4 + 3] + b[c4 + 3], 0.f);
  ((float4*)Y)[i4] = o;
}

// ---------------- host ----------------
extern "C" void kernel_launch(void* const* d_in, const int* in_sizes, int n_in,
                              void* d_out, int out_size, void* d_ws, size_t ws_size,
                              hipStream_t stream) {
  (void)in_sizes; (void)n_in; (void)out_size; (void)ws_size;
  const float* x = (const float*)d_in[0];
  // params flattened in dict insertion order, 10 leaves per layer:
  // 0 gcn_W, 1 gcn_b, 2 bn_g, 3 bn_b, 4 mlp_W1, 5 mlp_b1, 6 ln_g, 7 ln_b, 8 mlp_W2, 9 mlp_b2
  auto P = [&](int l, int i) -> const float* { return (const float*)d_in[1 + l * 10 + i]; };
  const int* ei = (const int*)d_in[1 + L * 10];
  const int* srcI = ei;
  const int* dstI = ei + E;

  char* w = (char*)d_ws;
  size_t off = 0;
  auto alloc = [&](size_t bytes) -> void* {
    off = (off + 255) & ~(size_t)255;
    void* p = w + off;
    off += bytes;
    return p;
  };
  float*    dinv  = (float*)alloc((size_t)N * 4);
  _Float16* wf    = (_Float16*)alloc((size_t)12 * 16384 * 2);
  float*    Hbuf  = (float*)alloc((size_t)N * C * 4);
  float*    W1buf = (float*)alloc((size_t)N * C * 4);
  float*    A1    = (float*)alloc((size_t)N * C * 4);
  float*    part  = (float*)alloc((size_t)RBLK * 256 * 4);
  float*    mu    = (float*)alloc(128 * 4);
  float*    istd  = (float*)alloc(128 * 4);

  // degrees -> dinv (in place)
  k_zero_f<<<(N + 255) / 256, 256, 0, stream>>>(dinv, N);
  k_deg<<<(E + 255) / 256, 256, 0, stream>>>(dstI, dinv);
  k_dinv<<<(N + 255) / 256, 256, 0, stream>>>(dinv);

  // fp32 weights -> pre-swizzled f16 B fragments
  for (int l = 0; l < L; ++l) {
    k_convw<<<64, 256, 0, stream>>>(P(l, 0), wf + (size_t)(l * 3 + 0) * 16384);
    k_convw<<<64, 256, 0, stream>>>(P(l, 4), wf + (size_t)(l * 3 + 1) * 16384);
    k_convw<<<64, 256, 0, stream>>>(P(l, 8), wf + (size_t)(l * 3 + 2) * 16384);
  }

  const int GB  = N / 16;                 // 6250 gemm blocks
  const int EWB = (N * (C / 4) + 255) / 256;
  const float* h = x;
  float* out = (float*)d_out;
  for (int l = 0; l < L; ++l) {
    k_gemm<0><<<GB, 256, 0, stream>>>(h, wf + (size_t)(l * 3 + 0) * 16384,
                                      nullptr, nullptr, nullptr, W1buf, C, 0);
    k_agg_init<<<EWB, 256, 0, stream>>>(W1buf, dinv, P(l, 1), A1);
    k_scatter<<<(E * 32) / 256, 256, 0, stream>>>(W1buf, dinv, srcI, dstI, A1);
    k_bn_partial<<<RBLK, 256, 0, stream>>>(A1, part);
    k_bn_finalize<<<1, 128, 0, stream>>>(part, mu, istd, RBLK);
    k_bn_apply<<<EWB, 256, 0, stream>>>(A1, mu, istd, P(l, 2), P(l, 3), Hbuf);
    k_gemm<1><<<GB, 256, 0, stream>>>(Hbuf, wf + (size_t)(l * 3 + 1) * 16384,
                                      P(l, 5), P(l, 6), P(l, 7), A1, C, 0);
    k_gemm<2><<<GB, 256, 0, stream>>>(A1, wf + (size_t)(l * 3 + 2) * 16384,
                                      P(l, 9), nullptr, nullptr, out, LC, l * C);
    h = Hbuf;
  }
}
